// ScaledDotProdAttention_48765058679197
// MI455X (gfx1250) — matrix operand order repaired and checked by reading, not compiled, not run on hardware
//
#include <hip/hip_runtime.h>

typedef __bf16 bf16_t;
typedef __attribute__((ext_vector_type(16))) __bf16 v16bf;
typedef __attribute__((ext_vector_type(8)))  __bf16 v8bf;
typedef __attribute__((ext_vector_type(8)))  float  v8f;
typedef __attribute__((ext_vector_type(4)))  float  v4f;

#define WMMA_BF16(A, B, C) \
    __builtin_amdgcn_wmma_f32_16x16x32_bf16(false, (A), false, (B), (short)0, (C), false, false)

// ---------- one-shot fp32 -> bf16 conversion of K and V into workspace ----------
__global__ __launch_bounds__(256)
void cvt_kv_kernel(const float* __restrict__ k, const float* __restrict__ v,
                   bf16_t* __restrict__ kws, bf16_t* __restrict__ vws)
{
    const size_t i = ((size_t)blockIdx.x * 256 + threadIdx.x) * 8;
    const v4f k0 = *(const v4f*)(k + i);
    const v4f k1 = *(const v4f*)(k + i + 4);
    const v4f v0 = *(const v4f*)(v + i);
    const v4f v1 = *(const v4f*)(v + i + 4);
    v8bf kb, vb;
#pragma unroll
    for (int u = 0; u < 4; ++u) {
        kb[u] = (bf16_t)k0[u]; kb[4 + u] = (bf16_t)k1[u];
        vb[u] = (bf16_t)v0[u]; vb[4 + u] = (bf16_t)v1[u];
    }
    *(v8bf*)(kws + i) = kb;
    *(v8bf*)(vws + i) = vb;
}

// ---------- async DMA helpers (GLOBAL_LOAD_ASYNC_TO_LDS, ASYNCcnt-tracked) ----------
__device__ __forceinline__ void async_b128(unsigned lds_addr, const bf16_t* g)
{
    asm volatile("global_load_async_to_lds_b128 %0, %1, off"
                 :: "v"(lds_addr), "v"(g) : "memory");
}
__device__ __forceinline__ void wait_async0()
{
    asm volatile("s_wait_asynccnt 0" ::: "memory");
}

// A-operand fragment from row-major bf16 LDS: two contiguous 8-half groups
__device__ __forceinline__ v16bf ldA(const bf16_t* p)
{
    const v8bf lo = *(const v8bf*)p;          // frag elements 0..7  (k = off .. off+7)
    const v8bf hi = *(const v8bf*)(p + 16);   // frag elements 8..15 (k = off+16 .. +23)
    return __builtin_shufflevector(lo, hi, 0,1,2,3,4,5,6,7,8,9,10,11,12,13,14,15);
}

__global__ __launch_bounds__(128)
void attn_flash_bf16_kernel(const float* __restrict__ q,
                            const bf16_t* __restrict__ kws,
                            const bf16_t* __restrict__ vws,
                            float* __restrict__ out)
{
    constexpr int N  = 2048;
    constexpr int DM = 1024;
    constexpr int D  = 64;
    const float LOG2E       = 1.4426950408889634f;
    const float scale2      = 0.125f * LOG2E;     // d^-0.5 folded with log2(e)
    const float CAUSAL_NEG2 = -50000.0f * LOG2E;  // causal mask, log2 domain
    const float DIAG_NEG2   = -10000.0f * LOG2E;  // shared-qk diag mask, log2 domain

    // double-buffered 64x64 bf16 tiles
    __shared__ __align__(16) bf16_t sK[2][64 * 64];
    __shared__ __align__(16) bf16_t sV[2][64 * 64];

    const int tid  = threadIdx.x;
    const int wave = tid >> 5;
    const int lane = tid & 31;
    const int ln16 = lane & 15;
    const bool hi  = lane >= 16;
    const int hioff = hi ? 8 : 0;

    const int h    = blockIdx.y;
    const int bi   = blockIdx.z;
    const int q0b  = blockIdx.x * 64;
    const int q0   = q0b + wave * 16;
    const int qrow = q0 + ln16;                   // this lane's q-row (S^T orientation)

    const size_t base = (size_t)bi * N * DM + (size_t)h * D;

    // ---- Q^T fragments (B layout: lane <-> q-row, elements <-> d), pre-scaled, once ----
    v16bf Qt[2];
#pragma unroll
    for (int c = 0; c < 2; ++c)
#pragma unroll
        for (int e = 0; e < 16; ++e)
            Qt[c][e] = (bf16_t)(q[base + (size_t)(q0 + ln16) * DM + c * 32 + hioff + ((e < 8) ? e : e + 8)] * scale2);

    v8f acc[4];
#pragma unroll
    for (int t = 0; t < 4; ++t) acc[t] = (v8f){};
    float mrow = -3.0e38f, lrow = 0.0f;

    // per-thread async-copy addressing: 1 row-half (32 bf16 = 64B = 4 x b128) per array
    const int jrow = tid >> 1;                    // 0..63
    const int dh   = (tid & 1) * 32;              // 0 or 32
    const unsigned lK[2] = { (unsigned)(uintptr_t)&sK[0][jrow * 64 + dh],
                             (unsigned)(uintptr_t)&sK[1][jrow * 64 + dh] };
    const unsigned lV[2] = { (unsigned)(uintptr_t)&sV[0][jrow * 64 + dh],
                             (unsigned)(uintptr_t)&sV[1][jrow * 64 + dh] };

    const int njb = q0b / 64 + 1;                 // causal: tiles up to block diagonal

    // prologue: start DMA of tile 0 into buffer 0
    {
        const bf16_t* gk = kws + base + (size_t)jrow * DM + dh;
        const bf16_t* gv = vws + base + (size_t)jrow * DM + dh;
#pragma unroll
        for (int i = 0; i < 4; ++i) {
            async_b128(lK[0] + i * 16, gk + i * 8);
            async_b128(lV[0] + i * 16, gv + i * 8);
        }
    }

    for (int jb = 0; jb < njb; ++jb) {
        const int j0  = jb * 64;
        const int cur = jb & 1;

        wait_async0();        // my copies for tile jb have landed
        __syncthreads();      // everyone's copies landed; prev buffer free

        // kick off DMA of next tile into the other buffer (overlaps compute below)
        if (jb + 1 < njb) {
            const bf16_t* gk = kws + base + (size_t)(j0 + 64 + jrow) * DM + dh;
            const bf16_t* gv = vws + base + (size_t)(j0 + 64 + jrow) * DM + dh;
#pragma unroll
            for (int i = 0; i < 4; ++i) {
                async_b128(lK[1 - cur] + i * 16, gk + i * 8);
                async_b128(lV[1 - cur] + i * 16, gv + i * 8);
            }
        }

        // ---- S^T = K (Q*scale)^T : four 16(j)x16(qrow) tiles ----
        v8f s[4];
#pragma unroll
        for (int nh = 0; nh < 4; ++nh) s[nh] = (v8f){};
#pragma unroll
        for (int c = 0; c < 2; ++c)
#pragma unroll
            for (int nh = 0; nh < 4; ++nh) {
                const v16bf a = ldA(&sK[cur][(nh * 16 + ln16) * 64 + c * 32 + hioff]);
                s[nh] = WMMA_BF16(a, Qt[c], s[nh]);
            }

        // ---- Masks: only the diagonal tile (wave-uniform branch) ----
        if (j0 + 63 >= q0) {
#pragma unroll
            for (int nh = 0; nh < 4; ++nh) {
                const int jh = j0 + nh * 16 + hioff;
#pragma unroll
                for (int r = 0; r < 8; ++r) {
                    const int jg = jh + r;
                    float x = s[nh][r];
                    x = (jg > qrow) ? CAUSAL_NEG2 : ((jg == qrow) ? DIAG_NEG2 : x);
                    s[nh][r] = x;
                }
            }
        }

        // ---- Online softmax, exp2 domain; lane-local + one cross-half shfl ----
        float mx = s[0][0];
#pragma unroll
        for (int nh = 0; nh < 4; ++nh)
#pragma unroll
            for (int r = 0; r < 8; ++r) mx = fmaxf(mx, s[nh][r]);
        mx = fmaxf(mx, __shfl_xor(mx, 16));

        const float mnew = fmaxf(mrow, mx);
        const float al   = __builtin_amdgcn_exp2f(mrow - mnew);

        float pe[32];
        float rs = 0.0f;
#pragma unroll
        for (int nh = 0; nh < 4; ++nh)
#pragma unroll
            for (int r = 0; r < 8; ++r) {
                const float p = __builtin_amdgcn_exp2f(s[nh][r] - mnew);
                pe[nh * 8 + r] = p;
                rs += p;
            }
        rs += __shfl_xor(rs, 16);
        lrow = lrow * al + rs;
        mrow = mnew;

        // ---- P in A layout: pure in-lane register permute (no LDS) ----
        v16bf Pf[2];
#pragma unroll
        for (int cc = 0; cc < 2; ++cc)
#pragma unroll
            for (int vv = 0; vv < 8; ++vv) {
                const int src = cc * 16 + ((vv < 4) ? 0 : 8) + (vv & 3) * 2;
                Pf[cc][2 * vv]     = (bf16_t)pe[src];
                Pf[cc][2 * vv + 1] = (bf16_t)pe[src + 1];
            }

        // ---- Broadcast alpha to O-row owners, rescale once, accumulate PV ----
        float alr[8];
#pragma unroll
        for (int r = 0; r < 8; ++r)
            alr[r] = __shfl(al, hioff + r);

#pragma unroll
        for (int t = 0; t < 4; ++t)
#pragma unroll
            for (int r = 0; r < 8; ++r) acc[t][r] *= alr[r];

#pragma unroll
        for (int cc = 0; cc < 2; ++cc)
#pragma unroll
            for (int t = 0; t < 4; ++t) {
                v16bf vf;
#pragma unroll
                for (int e = 0; e < 16; ++e)
                    vf[e] = sV[cur][(cc * 32 + hioff + ((e < 8) ? e : e + 8)) * 64 + t * 16 + ln16];
                acc[t] = WMMA_BF16(Pf[cc], vf, acc[t]);
            }
    }

    // ---- Normalize (broadcast 1/l per row) and write out ----
    const float linv = 1.0f / lrow;
#pragma unroll
    for (int r = 0; r < 8; ++r) {
        const float li  = __shfl(linv, hioff + r);
        const int   row = q0 + hioff + r;
        float* orow = out + base + (size_t)row * DM;
#pragma unroll
        for (int t = 0; t < 4; ++t)
            orow[t * 16 + ln16] = acc[t][r] * li;
    }
}

extern "C" void kernel_launch(void* const* d_in, const int* in_sizes, int n_in,
                              void* d_out, int out_size, void* d_ws, size_t ws_size,
                              hipStream_t stream) {
    (void)in_sizes; (void)n_in; (void)out_size; (void)ws_size;
    const float* q = (const float*)d_in[0];
    const float* k = (const float*)d_in[1];
    const float* v = (const float*)d_in[2];
    float* out = (float*)d_out;

    constexpr size_t NELEM = (size_t)2 * 2048 * 1024;   // per tensor
    bf16_t* kws = (bf16_t*)d_ws;
    bf16_t* vws = kws + NELEM;

    // pass 1: bf16 conversion of K,V (bandwidth-bound)
    cvt_kv_kernel<<<dim3((unsigned)(NELEM / 8 / 256)), dim3(256), 0, stream>>>(k, v, kws, vws);

    // pass 2: flash attention (compute-bound)
    dim3 grid(2048 / 64, 16, 2);   // (q-row tiles, heads, batch)
    attn_flash_bf16_kernel<<<grid, dim3(128), 0, stream>>>(q, kws, vws, out);
}
